// CycleTime_11647951307162
// MI455X (gfx1250) — matrix operand order repaired and checked by reading, not compiled, not run on hardware
//
#include <hip/hip_runtime.h>

typedef __attribute__((ext_vector_type(16))) _Float16 v16h;
typedef __attribute__((ext_vector_type(8)))  float    v8f;

#define B_ 8
#define C_ 512
#define T_ 4
#define H_ 30
#define W_ 30
#define R_ 12
#define P_ 25

// ---------------------------------------------------------------------------
// Kernel 1a: video pack.  relu + L2-normalize over C, f32 -> f16,
// layout [b][t][y][x(32 slots)][c(512)] with NATURAL channel order
// (this code's contiguous WMMA k order for A and B: lanes 0-15 K=0..15,
//  lanes 16-31 K=16..31, sequential within lane).
// One wave per (b,t,y,x) position; lane l handles channels l, l+32, ...
// ---------------------------------------------------------------------------
__global__ __launch_bounds__(256) void pack_video_k(const float* __restrict__ in,
                                                    _Float16* __restrict__ out) {
    int wave = (int)((blockIdx.x * blockDim.x + threadIdx.x) >> 5);
    int lane = threadIdx.x & 31;
    if (wave >= B_ * T_ * H_ * W_) return;
    int x = wave % W_;  int tmp = wave / W_;
    int y = tmp % H_;   tmp /= H_;
    int t = tmp % T_;   int b = tmp / T_;

    // input index: (b*512 + c)*3600 + t*900 + y*30 + x
    size_t base = (size_t)b * C_ * (T_ * H_ * W_) + (size_t)t * (H_ * W_) + y * W_ + x;

    float v[16];
    float ss = 0.f;
#pragma unroll
    for (int i = 0; i < 16; i++) {
        int c = i * 32 + lane;
        float f = in[base + (size_t)c * (T_ * H_ * W_)];
        f = fmaxf(f, 0.f);
        v[i] = f;
        ss += f * f;
    }
#pragma unroll
    for (int off = 16; off; off >>= 1) ss += __shfl_xor(ss, off, 32);
    float inv = 1.0f / fmaxf(sqrtf(ss), 1e-12f);

    size_t obase = (((size_t)(b * T_ + t) * H_ + y) * 32 + x) * C_;
#pragma unroll
    for (int i = 0; i < 16; i++)
        out[obase + i * 32 + lane] = (_Float16)(v[i] * inv);
}

// ---------------------------------------------------------------------------
// Kernel 1b: patch pack. Same normalize and the same NATURAL channel order,
// layout [b][y][x(32 slots)][c(512)]; no swizzle:
// memory slots = [c0..c31], so lane L<16 loads
// K=0..15 and lane L>=16 loads K=16..31 as one 32B chunk.
// ---------------------------------------------------------------------------
__global__ __launch_bounds__(256) void pack_patch_k(const float* __restrict__ in,
                                                    _Float16* __restrict__ out) {
    int wave = (int)((blockIdx.x * blockDim.x + threadIdx.x) >> 5);
    int lane = threadIdx.x & 31;
    if (wave >= B_ * H_ * W_) return;
    int x = wave % W_;  int tmp = wave / W_;
    int y = tmp % H_;   int b = tmp / H_;

    size_t base = (size_t)b * C_ * (H_ * W_) + (size_t)y * W_ + x;

    float v[16];
    float ss = 0.f;
#pragma unroll
    for (int i = 0; i < 16; i++) {
        int c = i * 32 + lane;
        float f = in[base + (size_t)c * (H_ * W_)];
        f = fmaxf(f, 0.f);
        v[i] = f;
        ss += f * f;
    }
#pragma unroll
    for (int off = 16; off; off >>= 1) ss += __shfl_xor(ss, off, 32);
    float inv = 1.0f / fmaxf(sqrtf(ss), 1e-12f);

    // natural slot for channel-within-block j = lane
    int j = lane;
    int f16slot = j;

    size_t obase = (((size_t)b * H_ + y) * 32 + x) * C_;
#pragma unroll
    for (int i = 0; i < 16; i++)
        out[obase + i * 32 + f16slot] = (_Float16)(v[i] * inv);
}

// ---------------------------------------------------------------------------
// Kernel 2: WMMA correlation, software-pipelined with a pinned schedule.
// grid = b(8) * xtile(2) * y(30) = 480 blocks, 256 threads (8 waves).
// A (16 queries x 512 ch, f16) stays in 128 VGPRs for the wave's lifetime.
// Per (t,dh) pair: 2 N-tiles (keys x'=0..31) x 16 K-steps = 32 WMMAs fed by
// an 8-fragment rotating B buffer (64 VGPRs).  sched_group_barrier pins the
// issue pattern [16 vmem-reads][24 x (1 wmma, 2 vmem-reads)][8 wmma] so 8
// fragment loads (16 global_load_b128) stay in flight ahead of the consuming
// WMMA instead of the scheduler's load->wait0->wmma serialization.
// Raw S tiles written to ws: [b][xt][y][pair(100)][Mrow(16)][x'(32)] f32.
// Pairs with y' out of range are skipped (kernel 3 treats them as logit 0).
// ---------------------------------------------------------------------------
__global__ __launch_bounds__(256)
__attribute__((amdgpu_waves_per_eu(2)))
void corr_wmma_k(const _Float16* __restrict__ packP,
                 const _Float16* __restrict__ packV,
                 float* __restrict__ S) {
    int wid  = threadIdx.x >> 5;
    int lane = threadIdx.x & 31;
    int y  = blockIdx.x % H_;
    int tq = blockIdx.x / H_;
    int xt = tq & 1;
    int b  = tq >> 1;

    int m  = lane & 15;
    int hi = lane >> 4;

    // A fragments: lane = (row m, K-half hi); contiguous 32B per lane
    // (natural pack order, k = 16*hi + s).
    const _Float16* aptr =
        packP + (((size_t)(b * H_ + y) * 32) + (xt * 16 + m)) * C_ + hi * 16;
    v16h a[16];
#pragma unroll
    for (int kf = 0; kf < 16; kf++)
        a[kf] = *(const v16h*)(aptr + kf * 32);

    for (int pair = wid; pair < 100; pair += 8) {
        int t  = pair / 25;
        int dh = pair % 25;
        int yy = y + dh - R_;
        if (yy < 0 || yy >= H_) continue;   // wave-uniform; logits stay 0

        // B: lane = (column n = m -> key x' = nt*16 + m, K-half hi), natural order
        const _Float16* bp0 =
            packV + (((size_t)((b * T_ + t) * H_ + yy) * 32) + m) * C_ + hi * 16;
        const _Float16* bp1 = bp0 + 16 * C_;   // keys x' = 16..31

        v8f acc0 = {};
        v8f acc1 = {};
        v16h bb[8];

        // prologue: fill the 8-deep fragment pipeline (ntile0, K 0..255)
#pragma unroll
        for (int i = 0; i < 8; i++)
            bb[i] = *(const v16h*)(bp0 + i * 32);

        // steady state: consume oldest fragment, refill slot with the next
        // one in the stream (wmma issued before the overwriting load)
#pragma unroll
        for (int kf = 0; kf < 8; kf++) {
            acc0 = __builtin_amdgcn_wmma_f32_16x16x32_f16(
                false, a[kf], false, bb[kf], (short)0, acc0, false, false);
            bb[kf] = *(const v16h*)(bp0 + (8 + kf) * 32);
        }
#pragma unroll
        for (int kf = 0; kf < 8; kf++) {
            acc0 = __builtin_amdgcn_wmma_f32_16x16x32_f16(
                false, a[8 + kf], false, bb[kf], (short)0, acc0, false, false);
            bb[kf] = *(const v16h*)(bp1 + kf * 32);
        }
#pragma unroll
        for (int kf = 0; kf < 8; kf++) {
            acc1 = __builtin_amdgcn_wmma_f32_16x16x32_f16(
                false, a[kf], false, bb[kf], (short)0, acc1, false, false);
            bb[kf] = *(const v16h*)(bp1 + (8 + kf) * 32);
        }
#pragma unroll
        for (int kf = 0; kf < 8; kf++) {
            acc1 = __builtin_amdgcn_wmma_f32_16x16x32_f16(
                false, a[8 + kf], false, bb[kf], (short)0, acc1, false, false);
        }

        // ---- pin the issue pattern for this scheduling region -------------
        // groups: 16 vmem-reads (prologue), then 24 x [1 wmma, 2 vmem-reads],
        // then the 8-wmma drain.  0x8 = MFMA/WMMA, 0x20 = VMEM read.
        __builtin_amdgcn_sched_group_barrier(0x020, 16, 0);
#pragma unroll
        for (int i = 0; i < 24; i++) {
            __builtin_amdgcn_sched_group_barrier(0x008, 1, 0);
            __builtin_amdgcn_sched_group_barrier(0x020, 2, 0);
        }
        __builtin_amdgcn_sched_group_barrier(0x008, 8, 0);
        // -------------------------------------------------------------------

        // C/D layout: lane L: N = L%16, rows M = r + 8*(L>=16)
        float* sp = S + ((((size_t)(b * 2 + xt) * H_ + y) * 100 + pair) * 512);
#pragma unroll
        for (int r = 0; r < 8; r++) {
            int Mr = r + 8 * hi;
            sp[Mr * 32 + m]      = acc0[r];
            sp[Mr * 32 + 16 + m] = acc1[r];
        }
    }
}

// ---------------------------------------------------------------------------
// Kernel 3: online softmax over the 625 offsets + validity mask.
// One lane per (b,t,y,x); 960 waves total; logits = 512 * S (or 0 if the
// shifted position is out of bounds -- exactly the reference's zero-padding),
// output -1.0 where invalid (same predicate).
// ---------------------------------------------------------------------------
__global__ __launch_bounds__(128) void softmax_mask_k(const float* __restrict__ S,
                                                      float* __restrict__ out) {
    int gw   = (int)((blockIdx.x * blockDim.x + threadIdx.x) >> 5);
    int lane = threadIdx.x & 31;
    if (gw >= B_ * T_ * H_) return;
    int y = gw % H_;  int tmp = gw / H_;
    int t = tmp % T_; int b = tmp / T_;
    int x = lane;
    if (x >= W_) return;

    int xt = x >> 4;
    int m  = x & 15;
    const float* sp =
        S + (((size_t)(b * 2 + xt) * H_ + y) * 100) * 512 + (size_t)m * 32;

    float Mx = -3.0e38f, Sm = 0.f;
    for (int dh = 0; dh < P_; dh++) {
        int  yy = y + dh - R_;
        bool rv = (yy >= 0) && (yy < H_);
        size_t pbase = (size_t)(t * 25 + dh) * 512;
        for (int dw = 0; dw < P_; dw++) {
            int  xq = x + dw - R_;
            bool cv = rv && (xq >= 0) && (xq < W_);
            float l = cv ? 512.0f * sp[pbase + xq] : 0.0f;
            float nM = fmaxf(Mx, l);
            Sm = Sm * __expf(Mx - nM) + __expf(l - nM);
            Mx = nM;
        }
    }
    float rS = 1.0f / Sm;

    // out[b][t][dh][dw][y][x]
    size_t obase = (size_t)(b * T_ + t) * (P_ * P_ * H_ * W_) + (size_t)y * W_ + x;
    for (int dh = 0; dh < P_; dh++) {
        int  yy = y + dh - R_;
        bool rv = (yy >= 0) && (yy < H_);
        size_t pbase = (size_t)(t * 25 + dh) * 512;
        for (int dw = 0; dw < P_; dw++) {
            int  xq = x + dw - R_;
            bool cv = rv && (xq >= 0) && (xq < W_);
            float l = cv ? 512.0f * sp[pbase + xq] : 0.0f;
            float p = __expf(l - Mx) * rS;
            out[obase + (size_t)(dh * P_ + dw) * (H_ * W_)] = cv ? p : -1.0f;
        }
    }
}

// ---------------------------------------------------------------------------
// launch
// ---------------------------------------------------------------------------
extern "C" void kernel_launch(void* const* d_in, const int* in_sizes, int n_in,
                              void* d_out, int out_size, void* d_ws, size_t ws_size,
                              hipStream_t stream) {
    const float* vin = (const float*)d_in[0];   // [8,512,4,30,30]
    const float* pin = (const float*)d_in[1];   // [8,512,30,30]
    float* out = (float*)d_out;                 // [8,4,25,25,30,30]
    char* ws = (char*)d_ws;

    // ws layout (all 1KB-aligned):
    //   packV : f16  8*4*30*32*512   = 31,457,280 B
    //   packP : f16  8*30*32*512     =  7,864,320 B (15,728,640 B reserved)
    //   Sbuf  : f32  8*2*30*100*512  = 98,304,000 B   (total ~145.5 MB)
    _Float16* packV = (_Float16*)ws;
    _Float16* packP = (_Float16*)(ws + (size_t)31457280);
    float*    Sbuf  = (float*)(ws + (size_t)31457280 + 15728640);

    pack_video_k<<<(B_ * T_ * H_ * W_) / 8, 256, 0, stream>>>(vin, packV);
    pack_patch_k<<<(B_ * H_ * W_) / 8, 256, 0, stream>>>(pin, packP);
    corr_wmma_k<<<B_ * 2 * H_, 256, 0, stream>>>(packP, packV, Sbuf);
    softmax_mask_k<<<(B_ * T_ * H_) / 4, 128, 0, stream>>>(Sbuf, out);
}
